// dualEncoder_52407190945939
// MI455X (gfx1250) — compile-verified, matrix operand order checked by reading, not run on hardware
//
#include <hip/hip_runtime.h>

// ---------------------------------------------------------------------------
// MI455X (gfx1250) dual-encoder block.
// v3: Q/K/V produced as bf16 directly from GEMM epilogues (halves attention
// traffic, staging becomes raw 16B copies); A-fragments preloaded before the
// K-loop so only the global B stream is waited on; temporal-attention Q/K
// staging uses gfx1250 async load-to-LDS (ASYNCcnt) with swizzled lane dests.
// ---------------------------------------------------------------------------

#define USE_ASYNC_LDS 1

typedef __bf16 bf16_t;
typedef __attribute__((ext_vector_type(16))) __bf16 v16bf;
typedef __attribute__((ext_vector_type(8)))  float  v8f;
typedef __attribute__((ext_vector_type(4)))  unsigned int u32x4;

#define Bb 8
#define Tt 64
#define Nn 256
#define Ff 128
static const long NTOT = (long)Bb * Tt * Nn * Ff;      // 16,777,216
static const int  NROWS = Bb * Tt * Nn;                // 131,072 rows of F=128

__device__ __forceinline__ int lane_id() { return threadIdx.x & 31; }

__device__ __forceinline__ v8f wmma_bf16(v16bf a, v16bf b, v8f c) {
  return __builtin_amdgcn_wmma_f32_16x16x32_bf16(false, a, false, b, (short)0, c,
                                                 false, false);
}

union FragU { u32x4 q[2]; v16bf v; };
union Pack8 { bf16_t h[8]; u32x4 u; };

// Convert 8 consecutive f32 -> 8 bf16 packed in one 16-byte quantity.
__device__ __forceinline__ u32x4 pack8(const float* __restrict__ src) {
  float4 a = *(const float4*)src;
  float4 b = *(const float4*)(src + 4);
  Pack8 p;
  p.h[0] = (bf16_t)a.x; p.h[1] = (bf16_t)a.y; p.h[2] = (bf16_t)a.z; p.h[3] = (bf16_t)a.w;
  p.h[4] = (bf16_t)b.x; p.h[5] = (bf16_t)b.y; p.h[6] = (bf16_t)b.z; p.h[7] = (bf16_t)b.w;
  return p.u;
}

// LDS fragment load (XOR-swizzled 16B chunks). Per CDNA5 ISA 7.12.2 the lane's
// 16 bf16 elements are rows r=(l&15), K = k0 + 8*(l>>4) + {0..7, 16..23}.
__device__ __forceinline__ v16bf frag_lds(const bf16_t* base, int ld, int k0,
                                          int mask) {
  int l = lane_id();
  int r = l & 15;
  int c0 = (k0 >> 3) + (l >> 4);
  const u32x4* rowp = (const u32x4*)(base + r * ld);
  FragU f;
  f.q[0] = rowp[c0 ^ (r & mask)];
  f.q[1] = rowp[(c0 + 2) ^ (r & mask)];
  return f.v;
}

// Global fragment load (unswizzled rows of length ld, 16B aligned).
__device__ __forceinline__ v16bf frag_g(const bf16_t* __restrict__ base, int ld,
                                        int k0) {
  int l = lane_id();
  const bf16_t* p = base + (l & 15) * ld + k0 + ((l >> 4) << 3);
  FragU f;
  f.q[0] = *(const u32x4*)p;
  f.q[1] = *(const u32x4*)(p + 16);
  return f.v;
}

#if USE_ASYNC_LDS
// gfx1250 async copy: 16B global -> LDS, tracked by ASYNCcnt (ISA 15.18.3).
__device__ __forceinline__ void async_copy16(const bf16_t* g, bf16_t* l) {
  unsigned lds = (unsigned)(unsigned long long)l;   // low 32b = LDS offset
  asm volatile("global_load_async_to_lds_b128 %0, %1, off"
               :: "v"(lds), "v"(g) : "memory");
}
__device__ __forceinline__ void async_wait0() {
  asm volatile("s_wait_asynccnt 0x0" ::: "memory");
}
#endif

// ---------------------------------------------------------------------------
// Elementwise kernels
// ---------------------------------------------------------------------------
__global__ void k_add(const float* __restrict__ a, const float* __restrict__ b,
                      float* __restrict__ c, long n) {
  long i = (long)blockIdx.x * 256 + threadIdx.x;
  if (i < n) c[i] = a[i] + b[i];
}

__global__ void k_add_bcast(const float* __restrict__ x, const float* __restrict__ adj,
                            float* __restrict__ y, long n) {
  long i = (long)blockIdx.x * 256 + threadIdx.x;
  if (i < n) y[i] = x[i] + adj[i & (Nn * Ff - 1)];
}

// LayerNorm (no affine) over F=128; one wave per row (wave32).
__global__ void k_ln(float* __restrict__ x, int nrows) {
  int row = blockIdx.x * 8 + (threadIdx.x >> 5);
  if (row >= nrows) return;
  int l = threadIdx.x & 31;
  float* p = x + (long)row * Ff;
  float v0 = p[l], v1 = p[l + 32], v2 = p[l + 64], v3 = p[l + 96];
  float s = v0 + v1 + v2 + v3;
  float q = v0 * v0 + v1 * v1 + v2 * v2 + v3 * v3;
#pragma unroll
  for (int m = 16; m >= 1; m >>= 1) {
    s += __shfl_xor(s, m, 32);
    q += __shfl_xor(q, m, 32);
  }
  float mean = s * (1.f / Ff);
  float var = q * (1.f / Ff) - mean * mean;
  float inv = rsqrtf(var + 1e-5f);
  p[l] = (v0 - mean) * inv;
  p[l + 32] = (v1 - mean) * inv;
  p[l + 64] = (v2 - mean) * inv;
  p[l + 96] = (v3 - mean) * inv;
}

// ---------------------------------------------------------------------------
// Weight packing
// ---------------------------------------------------------------------------
__global__ void k_pack_wt(const float* __restrict__ W, bf16_t* __restrict__ WT) {
  int idx = blockIdx.x * 256 + threadIdx.x;     // 16384
  int o = idx >> 7, i = idx & 127;
  WT[idx] = (bf16_t)W[i * 128 + o];
}

__global__ void k_pack_convw(const float* __restrict__ w, bf16_t* __restrict__ WCt) {
  int idx = blockIdx.x * 256 + threadIdx.x;     // 32768
  int o = idx >> 8, kr = idx & 255;
  int half = kr >> 7, ii = kr & 127;
  WCt[idx] = (bf16_t)w[(o * 128 + ii) * 2 + half];
}

// V[bt][n][f] bf16 -> VT[bt][f][n] bf16 (for att@V B-fragments)
__global__ void k_vt(const bf16_t* __restrict__ V, bf16_t* __restrict__ VT, long n) {
  long idx = (long)blockIdx.x * 256 + threadIdx.x;
  if (idx >= n) return;
  long bt = idx >> 15;
  int r = (int)(idx & 32767);
  int f = r >> 8, nn = r & 255;
  VT[idx] = V[bt * 32768 + (long)nn * 128 + f];
}

// ---------------------------------------------------------------------------
// Linear kernels: out = A[M,128] @ W + bias. A-frags preloaded; B from global.
// ---------------------------------------------------------------------------
__global__ void k_gemm_lin(const float* __restrict__ A, const bf16_t* __restrict__ WT,
                           const float* __restrict__ bias, const float* __restrict__ res,
                           float* __restrict__ out, int relu) {
  __shared__ __align__(16) bf16_t sA[16 * 128];
  int tid = threadIdx.x;
  long rowBase = (long)blockIdx.x * 16;
  const float* Ab = A + rowBase * Ff;
  {
    int r = tid >> 4;
    int cc = tid & 15;
    ((u32x4*)(sA + r * 128))[cc ^ (r & 7)] = pack8(Ab + r * 128 + cc * 8);
  }
  if (res) __builtin_prefetch(res + rowBase * Ff + tid * 8, 0, 0);
  __syncthreads();
  int n0 = (tid >> 5) * 16;
  const bf16_t* Wb = WT + n0 * 128;
  v16bf a0 = frag_lds(sA, 128, 0, 7);
  v16bf a1 = frag_lds(sA, 128, 32, 7);
  v16bf a2 = frag_lds(sA, 128, 64, 7);
  v16bf a3 = frag_lds(sA, 128, 96, 7);
  v8f c = {};
  c = wmma_bf16(a0, frag_g(Wb, 128, 0), c);
  c = wmma_bf16(a1, frag_g(Wb, 128, 32), c);
  c = wmma_bf16(a2, frag_g(Wb, 128, 64), c);
  c = wmma_bf16(a3, frag_g(Wb, 128, 96), c);
  int l = lane_id();
  int col = n0 + (l & 15);
  int ro = (l >> 4) << 3;
  float bi = bias[col];
#pragma unroll
  for (int r = 0; r < 8; ++r) {
    long row = rowBase + ro + r;
    float v = c[r] + bi;
    if (res) v += res[row * Ff + col];
    if (relu) v = v > 0.f ? v : 0.f;
    out[row * Ff + col] = v;
  }
}

// Variant with bf16 output (for Q/K/V): halves attention-side traffic.
__global__ void k_gemm_lin_bf16(const float* __restrict__ A,
                                const bf16_t* __restrict__ WT,
                                const float* __restrict__ bias,
                                bf16_t* __restrict__ out) {
  __shared__ __align__(16) bf16_t sA[16 * 128];
  int tid = threadIdx.x;
  long rowBase = (long)blockIdx.x * 16;
  const float* Ab = A + rowBase * Ff;
  {
    int r = tid >> 4;
    int cc = tid & 15;
    ((u32x4*)(sA + r * 128))[cc ^ (r & 7)] = pack8(Ab + r * 128 + cc * 8);
  }
  __syncthreads();
  int n0 = (tid >> 5) * 16;
  const bf16_t* Wb = WT + n0 * 128;
  v16bf a0 = frag_lds(sA, 128, 0, 7);
  v16bf a1 = frag_lds(sA, 128, 32, 7);
  v16bf a2 = frag_lds(sA, 128, 64, 7);
  v16bf a3 = frag_lds(sA, 128, 96, 7);
  v8f c = {};
  c = wmma_bf16(a0, frag_g(Wb, 128, 0), c);
  c = wmma_bf16(a1, frag_g(Wb, 128, 32), c);
  c = wmma_bf16(a2, frag_g(Wb, 128, 64), c);
  c = wmma_bf16(a3, frag_g(Wb, 128, 96), c);
  int l = lane_id();
  int col = n0 + (l & 15);
  int ro = (l >> 4) << 3;
  float bi = bias[col];
#pragma unroll
  for (int r = 0; r < 8; ++r) {
    long row = rowBase + ro + r;
    out[row * Ff + col] = (bf16_t)(c[r] + bi);
  }
}

// ---------------------------------------------------------------------------
// Fused temporal attention per (b,n): S = QK^T/4, causal softmax, O = P V.
// Q,K,V are bf16. grid = B*N = 2048, block 256. 64 KB LDS.
// ---------------------------------------------------------------------------
__global__ void k_temporal_attn(const bf16_t* __restrict__ Q,
                                const bf16_t* __restrict__ K,
                                const bf16_t* __restrict__ V,
                                float* __restrict__ O) {
  __shared__ __align__(16) char smem[65536];
  bf16_t* sQ  = (bf16_t*)smem;             // [64][128] 16KB (dead after S)
  bf16_t* sK  = (bf16_t*)(smem + 16384);   // [64][128] 16KB
  bf16_t* sVt = (bf16_t*)(smem + 32768);   // [128][64] 16KB (transposed V)
  float*  sS  = (float*)(smem + 49152);    // [64][64]  16KB
  bf16_t* sP  = (bf16_t*)smem;             // [64][64] bf16, overlays sQ

  int bn = blockIdx.x;
  int b = bn >> 8;
  int n = bn & 255;
  long base = ((long)b * Tt * Nn + n) * Ff;
  const long tstride = (long)Nn * Ff;
  int tid = threadIdx.x;

  // stage Q,K: raw 16B copies (bf16), swizzled LDS destinations
  for (int ci = tid; ci < 1024; ci += 256) {
    int t = ci >> 4;
    int cc = ci & 15;
    long g = base + (long)t * tstride + cc * 8;
    bf16_t* dq = sQ + t * 128 + ((cc ^ (t & 7)) << 3);
    bf16_t* dk = sK + t * 128 + ((cc ^ (t & 7)) << 3);
#if USE_ASYNC_LDS
    async_copy16(Q + g, dq);
    async_copy16(K + g, dk);
#else
    *(u32x4*)dq = *(const u32x4*)(Q + g);
    *(u32x4*)dk = *(const u32x4*)(K + g);
#endif
  }
  // stage V transposed: sVt[f][t], swizzled scalar writes
  for (int i = tid; i < 8192; i += 256) {
    int t = i >> 7, f = i & 127;
    sVt[f * 64 + (((t >> 3) ^ (f & 7)) << 3) + (t & 7)] =
        V[base + (long)t * tstride + f];
  }
#if USE_ASYNC_LDS
  async_wait0();
#endif
  __syncthreads();

  int wave = tid >> 5;
  // S = Q K^T : 16 tiles of 16x16, 2 per wave. B(k=f,n=t_key)=K[t_key][f].
  for (int tile = wave; tile < 16; tile += 8) {
    int mt = tile >> 2, nt = tile & 3;
    v8f c = {};
#pragma unroll
    for (int kk = 0; kk < 4; ++kk) {
      v16bf a = frag_lds(sQ + mt * 16 * 128, 128, kk * 32, 7);
      v16bf bb = frag_lds(sK + nt * 16 * 128, 128, kk * 32, 7);
      c = wmma_bf16(a, bb, c);
    }
    int l = lane_id();
    int col = nt * 16 + (l & 15);
    int ro = (l >> 4) << 3;
#pragma unroll
    for (int r = 0; r < 8; ++r) sS[(mt * 16 + ro + r) * 64 + col] = c[r];
  }
  __syncthreads();

  // causal softmax per row (scale 1/sqrt(D)=0.25); emit bf16 P (swizzled)
  if (tid < 64) {
    int t = tid;
    float mx = -3.4e38f;
    for (int j = 0; j <= t; ++j) {
      float v = sS[t * 64 + j] * 0.25f;
      sS[t * 64 + j] = v;
      if (v > mx) mx = v;
    }
    float sum = 0.f;
    for (int j = 0; j <= t; ++j) {
      float e = __expf(sS[t * 64 + j] - mx);
      sS[t * 64 + j] = e;
      sum += e;
    }
    float inv = 1.f / sum;
    for (int j = 0; j < 64; ++j) {
      float pv = (j <= t) ? sS[t * 64 + j] * inv : 0.f;
      sP[t * 64 + (((j >> 3) ^ (t & 7)) << 3) + (j & 7)] = (bf16_t)pv;
    }
  }
  __syncthreads();

  // O = P V : 32 tiles (4 row x 8 col), 4 per wave. B(k=t,n=f)=sVt[f][t].
  for (int tile = wave; tile < 32; tile += 8) {
    int mt = tile >> 3, nt = tile & 7;
    v8f c = {};
#pragma unroll
    for (int kk = 0; kk < 2; ++kk) {
      v16bf a = frag_lds(sP + mt * 16 * 64, 64, kk * 32, 7);
      v16bf bb = frag_lds(sVt + nt * 16 * 64, 64, kk * 32, 7);
      c = wmma_bf16(a, bb, c);
    }
    int l = lane_id();
    int col = nt * 16 + (l & 15);
    int ro = (l >> 4) << 3;
#pragma unroll
    for (int r = 0; r < 8; ++r) {
      int t = mt * 16 + ro + r;
      O[base + (long)t * tstride + col] = c[r];
    }
  }
}

// ---------------------------------------------------------------------------
// Spatial S = Q K^T / 4 per (b,t): [256,256], K=128. Q,K bf16.
// ---------------------------------------------------------------------------
__global__ void k_qk_spatial(const bf16_t* __restrict__ Q, const bf16_t* __restrict__ K,
                             float* __restrict__ S) {
  __shared__ __align__(16) bf16_t sQ[16 * 128];
  __shared__ __align__(16) bf16_t sK[256 * 32];
  int bt = blockIdx.x >> 4;
  int mt = blockIdx.x & 15;
  const bf16_t* Qb = Q + (long)bt * Nn * Ff + mt * 16 * Ff;
  const bf16_t* Kb = K + (long)bt * Nn * Ff;
  float* Sb = S + (long)bt * Nn * Nn + (long)mt * 16 * Nn;
  int tid = threadIdx.x;
  {
    int r = tid >> 4, cc = tid & 15;
    *(u32x4*)(sQ + r * 128 + ((cc ^ (r & 7)) << 3)) =
        *(const u32x4*)(Qb + r * 128 + cc * 8);
  }
  int wave = tid >> 5;
  v8f c0 = {}, c1 = {};
  for (int kk = 0; kk < 4; ++kk) {
    __syncthreads();
    for (int ci = tid; ci < 1024; ci += 256) {       // 256 keys x 4 chunks
      int nk = ci >> 2, cc = ci & 3;
      *(u32x4*)(sK + nk * 32 + ((cc ^ (nk & 3)) << 3)) =
          *(const u32x4*)(Kb + nk * 128 + kk * 32 + cc * 8);
    }
    __syncthreads();
    v16bf a = frag_lds(sQ, 128, kk * 32, 7);
    v16bf b0 = frag_lds(sK + (wave * 16) * 32, 32, 0, 3);
    v16bf b1 = frag_lds(sK + ((wave + 8) * 16) * 32, 32, 0, 3);
    c0 = wmma_bf16(a, b0, c0);
    c1 = wmma_bf16(a, b1, c1);
  }
  int l = lane_id();
  int cl = l & 15;
  int ro = (l >> 4) << 3;
#pragma unroll
  for (int r = 0; r < 8; ++r) {
    Sb[(ro + r) * Nn + wave * 16 + cl] = c0[r] * 0.25f;
    Sb[(ro + r) * Nn + (wave + 8) * 16 + cl] = c1[r] * 0.25f;
  }
}

// Row softmax over 256 entries; one wave per row.
__global__ void k_softmax_rows(float* __restrict__ att, int nrows) {
  int row = blockIdx.x * 8 + (threadIdx.x >> 5);
  if (row >= nrows) return;
  int l = threadIdx.x & 31;
  float* p = att + (long)row * 256;
  float v[8];
  float mx = -3.4e38f;
#pragma unroll
  for (int j = 0; j < 8; ++j) { v[j] = p[l + j * 32]; mx = fmaxf(mx, v[j]); }
#pragma unroll
  for (int m = 16; m >= 1; m >>= 1) mx = fmaxf(mx, __shfl_xor(mx, m, 32));
  float s = 0.f;
#pragma unroll
  for (int j = 0; j < 8; ++j) { v[j] = __expf(v[j] - mx); s += v[j]; }
#pragma unroll
  for (int m = 16; m >= 1; m >>= 1) s += __shfl_xor(s, m, 32);
  float inv = 1.f / s;
#pragma unroll
  for (int j = 0; j < 8; ++j) p[l + j * 32] = v[j] * inv;
}

// cp[bt,n] = argmax over query rows of att[bt,:,n] (first max wins).
__global__ void k_colargmax(const float* __restrict__ att, int* __restrict__ cp,
                            int total) {
  int idx = blockIdx.x * 256 + threadIdx.x;
  if (idx >= total) return;
  int bt = idx >> 8, n = idx & 255;
  const float* p = att + (long)bt * Nn * Nn + n;
  float best = p[0];
  int bi = 0;
  for (int q = 1; q < 256; ++q) {
    float v = p[(long)q * 256];
    if (v > best) { best = v; bi = q; }
  }
  cp[idx] = bi;
}

// av = att @ V per (b,t): [256,128], K=256. B from global VT[bt][f][n] bf16.
__global__ void k_av_spatial(const float* __restrict__ att, const bf16_t* __restrict__ VT,
                             float* __restrict__ AV) {
  __shared__ __align__(16) bf16_t sA[16 * 256];
  int bt = blockIdx.x >> 4;
  int mt = blockIdx.x & 15;
  const float* Ab = att + (long)bt * 65536 + (long)mt * 16 * 256;
  const bf16_t* VTb = VT + (long)bt * 32768;
  float* Ob = AV + (long)bt * 32768 + mt * 16 * 128;
  int tid = threadIdx.x;
  for (int ci = tid; ci < 512; ci += 256) {          // 16 rows x 32 chunks
    int r = ci >> 5, cc = ci & 31;
    ((u32x4*)(sA + r * 256))[cc ^ (r & 7)] = pack8(Ab + r * 256 + cc * 8);
  }
  __syncthreads();
  int n0 = (tid >> 5) * 16;
  const bf16_t* Wb = VTb + n0 * 256;
  v16bf a0 = frag_lds(sA, 256, 0, 7);
  v16bf a1 = frag_lds(sA, 256, 32, 7);
  v16bf a2 = frag_lds(sA, 256, 64, 7);
  v16bf a3 = frag_lds(sA, 256, 96, 7);
  v16bf a4 = frag_lds(sA, 256, 128, 7);
  v16bf a5 = frag_lds(sA, 256, 160, 7);
  v16bf a6 = frag_lds(sA, 256, 192, 7);
  v16bf a7 = frag_lds(sA, 256, 224, 7);
  v8f acc = {};
  acc = wmma_bf16(a0, frag_g(Wb, 256, 0), acc);
  acc = wmma_bf16(a1, frag_g(Wb, 256, 32), acc);
  acc = wmma_bf16(a2, frag_g(Wb, 256, 64), acc);
  acc = wmma_bf16(a3, frag_g(Wb, 256, 96), acc);
  acc = wmma_bf16(a4, frag_g(Wb, 256, 128), acc);
  acc = wmma_bf16(a5, frag_g(Wb, 256, 160), acc);
  acc = wmma_bf16(a6, frag_g(Wb, 256, 192), acc);
  acc = wmma_bf16(a7, frag_g(Wb, 256, 224), acc);
  int l = lane_id();
  int col = n0 + (l & 15);
  int ro = (l >> 4) << 3;
#pragma unroll
  for (int r = 0; r < 8; ++r) Ob[(ro + r) * Ff + col] = acc[r];
}

// val[row,:] = av[bt, cp[row], :]
__global__ void k_gather_rows(const float* __restrict__ AV, const int* __restrict__ cp,
                              float* __restrict__ OUT) {
  int row = blockIdx.x;
  int f = threadIdx.x;
  int bt = row >> 8;
  int src = cp[row];
  OUT[(long)row * Ff + f] = AV[((long)bt * Nn + src) * Ff + f];
}

// y = relu([x(t-1), x(t)] @ Wc + bias). K=256, B from global WCt[o][krow].
__global__ void k_tcn(const float* __restrict__ X, const bf16_t* __restrict__ WCt,
                      const float* __restrict__ bias, float* __restrict__ Y) {
  __shared__ __align__(16) bf16_t sA[16 * 256];
  int tid = threadIdx.x;
  long rowBase = (long)blockIdx.x * 16;
  for (int ci = tid; ci < 512; ci += 256) {          // 16 rows x 32 chunks
    int r = ci >> 5;
    int cc = ci & 31;
    long grow = rowBase + r;
    int t = (int)((grow >> 8) & (Tt - 1));
    u32x4 u;
    if (cc < 16) {                                   // tap for x[t-1]
      if (t == 0) { u32x4 z = {0u, 0u, 0u, 0u}; u = z; }
      else u = pack8(X + (grow - Nn) * Ff + cc * 8);
    } else {
      u = pack8(X + grow * Ff + (cc - 16) * 8);
    }
    ((u32x4*)(sA + r * 256))[cc ^ (r & 7)] = u;
  }
  __syncthreads();
  int n0 = (tid >> 5) * 16;
  const bf16_t* Wb = WCt + n0 * 256;
  v8f acc = {};
#pragma unroll
  for (int kk = 0; kk < 8; ++kk) {
    v16bf a = frag_lds(sA, 256, kk * 32, 7);
    v16bf b = frag_g(Wb, 256, kk * 32);
    acc = wmma_bf16(a, b, acc);
  }
  int l = lane_id();
  int col = n0 + (l & 15);
  int ro = (l >> 4) << 3;
  float bi = bias[col];
#pragma unroll
  for (int r = 0; r < 8; ++r) {
    long row = rowBase + ro + r;
    float v = acc[r] + bi;
    Y[row * Ff + col] = v > 0.f ? v : 0.f;
  }
}

// ---------------------------------------------------------------------------
// Host orchestration
// ---------------------------------------------------------------------------
struct AttnP {
  const float *qw, *kw, *vw, *ow, *f1w, *f2w;
  const float *qb, *kb, *vb, *ob, *f1b, *f2b;
};

static AttnP attn_from(void* const* d_in, int base) {
  AttnP p;
  p.qw = (const float*)d_in[base + 0];
  p.kw = (const float*)d_in[base + 1];
  p.vw = (const float*)d_in[base + 2];
  p.ow = (const float*)d_in[base + 3];
  p.f1w = (const float*)d_in[base + 4];
  p.f2w = (const float*)d_in[base + 5];
  p.qb = (const float*)d_in[base + 6];
  p.kb = (const float*)d_in[base + 7];
  p.vb = (const float*)d_in[base + 8];
  p.ob = (const float*)d_in[base + 9];
  p.f1b = (const float*)d_in[base + 10];
  p.f2b = (const float*)d_in[base + 11];
  return p;
}

extern "C" void kernel_launch(void* const* d_in, const int* in_sizes, int n_in,
                              void* d_out, int out_size, void* d_ws, size_t ws_size,
                              hipStream_t stream) {
  (void)in_sizes; (void)n_in; (void)out_size; (void)ws_size;
  const float* xl = (const float*)d_in[0];
  const float* xh = (const float*)d_in[1];
  const float* te = (const float*)d_in[2];
  const float* adjgat = (const float*)d_in[3];
  AttnP tatt = attn_from(d_in, 4);
  AttnP ssal = attn_from(d_in, 16);
  AttnP ssah = attn_from(d_in, 28);
  const float* conv_w = (const float*)d_in[40];
  const float* conv_b = (const float*)d_in[41];

  float* ws = (float*)d_ws;
  float* X   = ws + 0 * NTOT;
  bf16_t* Qb = (bf16_t*)(ws + 1 * NTOT);   // bf16 Q (NTOT elems = NTOT*2 bytes)
  bf16_t* Kb = (bf16_t*)(ws + 2 * NTOT);
  bf16_t* Vb = (bf16_t*)(ws + 3 * NTOT);
  float* Ob  = ws + 4 * NTOT;   // attn out / av
  float* Yb  = ws + 5 * NTOT;
  float* Hb  = ws + 6 * NTOT;
  float* XLT = ws + 7 * NTOT;   // temporal branch output
  float* XHT = ws + 8 * NTOT;   // tcn branch output
  float* XS  = ws + 9 * NTOT;   // x + adjgat
  float* VAL = ws + 10 * NTOT;
  float* ATT = ws + 11 * NTOT;  // 2*NTOT floats: [B*T,256,256]
  int*   CP  = (int*)(ws + 13 * NTOT);
  bf16_t* WTS = (bf16_t*)(CP + NROWS);      // 18 packed weights, 16384 each
  bf16_t* WCt = WTS + 18 * 16384;           // 32768
  bf16_t* VT  = WCt + 32768;                // NTOT bf16

  float* out0 = (float*)d_out;
  float* out1 = out0 + NTOT;

  const int GE = (int)(NTOT / 256);
  const int GM = NROWS / 16;                // 8192
  const int GLN = NROWS / 8;                // 16384

  // ---- pack all weights (transposed bf16) ----
  const float* wsrc[18] = {tatt.qw, tatt.kw, tatt.vw, tatt.ow, tatt.f1w, tatt.f2w,
                           ssal.qw, ssal.kw, ssal.vw, ssal.ow, ssal.f1w, ssal.f2w,
                           ssah.qw, ssah.kw, ssah.vw, ssah.ow, ssah.f1w, ssah.f2w};
  for (int i = 0; i < 18; ++i)
    k_pack_wt<<<64, 256, 0, stream>>>(wsrc[i], WTS + (long)i * 16384);
  k_pack_convw<<<128, 256, 0, stream>>>(conv_w, WCt);

  auto gemm = [&](const float* A, const bf16_t* WT, const float* bias,
                  const float* res, float* out, int relu) {
    k_gemm_lin<<<GM, 256, 0, stream>>>(A, WT, bias, res, out, relu);
  };
  auto gemm_bf = [&](const float* A, const bf16_t* WT, const float* bias,
                     bf16_t* out) {
    k_gemm_lin_bf16<<<GM, 256, 0, stream>>>(A, WT, bias, out);
  };
  auto WTp = [&](int i) { return WTS + (long)i * 16384; };

  // ---- temporal attention on xl ----
  k_add<<<GE, 256, 0, stream>>>(xl, te, X, NTOT);
  gemm_bf(X, WTp(0), tatt.qb, Qb);
  gemm_bf(X, WTp(1), tatt.kb, Kb);
  gemm_bf(X, WTp(2), tatt.vb, Vb);
  k_temporal_attn<<<Bb * Nn, 256, 0, stream>>>(Qb, Kb, Vb, Ob);
  gemm(Ob, WTp(3), tatt.ob, X, Yb, 0);
  k_ln<<<GLN, 256, 0, stream>>>(Yb, NROWS);
  gemm(Yb, WTp(4), tatt.f1b, nullptr, Hb, 1);
  gemm(Hb, WTp(5), tatt.f2b, Yb, XLT, 0);
  k_ln<<<GLN, 256, 0, stream>>>(XLT, NROWS);

  // ---- TCN on xh ----
  k_tcn<<<GM, 256, 0, stream>>>(xh, WCt, conv_b, XHT);

  // ---- sparse spatial attention (shared scratch, run twice) ----
  auto spatial = [&](const float* input, const AttnP& P, int wbase, float* outp) {
    k_add_bcast<<<GE, 256, 0, stream>>>(input, adjgat, XS, NTOT);
    gemm_bf(XS, WTp(wbase + 0), P.qb, Qb);
    gemm_bf(XS, WTp(wbase + 1), P.kb, Kb);
    gemm_bf(XS, WTp(wbase + 2), P.vb, Vb);
    k_qk_spatial<<<Bb * Tt * 16, 256, 0, stream>>>(Qb, Kb, ATT);
    k_softmax_rows<<<GLN, 256, 0, stream>>>(ATT, NROWS);
    k_colargmax<<<NROWS / 256, 256, 0, stream>>>(ATT, CP, NROWS);
    k_vt<<<GE, 256, 0, stream>>>(Vb, VT, NTOT);
    k_av_spatial<<<Bb * Tt * 16, 256, 0, stream>>>(ATT, VT, Ob);
    k_gather_rows<<<NROWS, 128, 0, stream>>>(Ob, CP, VAL);
    gemm(VAL, WTp(wbase + 3), P.ob, XS, Yb, 0);
    k_ln<<<GLN, 256, 0, stream>>>(Yb, NROWS);
    gemm(Yb, WTp(wbase + 4), P.f1b, nullptr, Hb, 1);
    gemm(Hb, WTp(wbase + 5), P.f2b, input, outp, 0);  // fused final residual
  };
  spatial(XLT, ssal, 6, out0);
  spatial(XHT, ssah, 12, out1);
}
